// SAM_44006234915540
// MI455X (gfx1250) — matrix operand order repaired and checked by reading, not compiled, not run on hardware
//
#include <hip/hip_runtime.h>

// ---------------------------------------------------------------------------
// Deformable conv for MI455X (gfx1250): bf16 WMMA GEMM formulation.
// Shapes: B=4, C=256, H=W=80, K=3, Cout=256.
// GEMM: M = 25600 pixels, N = 256, K = 2304 (c*9+kk).
// Round 2: MTILE 32 -> 64 (halves L2 B-traffic, doubles WMMA density),
//          K-chunk 576 -> 288 to stay under 64 KB static LDS.
// ---------------------------------------------------------------------------

typedef __attribute__((ext_vector_type(16))) __bf16 bf16x16;
typedef __attribute__((ext_vector_type(8)))  __bf16 bf16x8;
typedef __attribute__((ext_vector_type(8)))  float  f32x8;

#define B_    4
#define C_    256
#define H_    80
#define W_    80
#define HW_   6400
#define KK_   9
#define KDIM_ 2304      // C_ * KK_
#define COUT_ 256
#define M_    25600     // B_ * HW_
#define MTILE 64        // pixels per workgroup
#define ENT_  (MTILE * KK_)   // 576 coord-table entries
#define CCH   32        // channels per K-chunk
#define KCH   288       // CCH * KK_
#define KPAD  296       // padded LDS row stride (bank spread, 16B-aligned rows)
#define NCHUNK 8        // C_ / CCH
#define KTL   9         // KCH / 32

// ---------------------------------------------------------------------------
// Kernel 1: NCHW -> NHWC transpose of x (so channel gathers are coalesced).
// ---------------------------------------------------------------------------
__global__ __launch_bounds__(256) void transpose_nchw_nhwc(
    const float* __restrict__ x, float* __restrict__ xt)
{
  __shared__ float tile[32][33];
  const int hw0 = blockIdx.x * 32;
  const int c0  = blockIdx.y * 32;
  const int b   = blockIdx.z;
  const int tx  = threadIdx.x;   // 0..31
  const int ty  = threadIdx.y;   // 0..7
  #pragma unroll
  for (int i = 0; i < 32; i += 8)
    tile[ty + i][tx] = x[(b * C_ + c0 + ty + i) * HW_ + hw0 + tx];
  __syncthreads();
  #pragma unroll
  for (int i = 0; i < 32; i += 8)
    xt[(b * HW_ + hw0 + ty + i) * C_ + c0 + tx] = tile[tx][ty + i];
}

// ---------------------------------------------------------------------------
// Kernel 2: pack w_deform (f32 [Cout][C][3][3]) into bf16 WMMA B-fragments.
// Fragment layout chosen here (contiguous k, same as A) for 16x16x32 bf16 WMMA (B is 32x16 K-major):
//   lane L -> column n = L%16 ; element i -> k = i + (L<16 ? 0 : 16).
// Flat index: (((kt*16 + nt)*32 + lane)*16 + i).
// ---------------------------------------------------------------------------
__global__ __launch_bounds__(256) void pack_bfrag(
    const float* __restrict__ wd, __bf16* __restrict__ bfrag)
{
  const int idx  = blockIdx.x * 256 + threadIdx.x;   // 0 .. 589823
  const int i    = idx & 15;
  const int lane = (idx >> 4) & 31;
  const int t    = idx >> 9;          // kt*16 + nt
  const int nt   = t & 15;
  const int kt   = t >> 4;
  const int n    = nt * 16 + (lane & 15);
  const int k    = kt * 32 + i + ((lane & 16) ? 16 : 0);
  const int c    = k / KK_;
  const int kk   = k % KK_;
  bfrag[idx] = (__bf16)wd[(n * C_ + c) * KK_ + kk];
}

// ---------------------------------------------------------------------------
// Kernel 3: offset conv (3x3, pad 1, C=256 -> 18). VALU; weight indices are
// thread-uniform -> scalar loads; x loads coalesced in NCHW.
// ---------------------------------------------------------------------------
__global__ __launch_bounds__(256) void offset_conv_kernel(
    const float* __restrict__ x, const float* __restrict__ w_off,
    float* __restrict__ offs)
{
  const int pix = blockIdx.x * 256 + threadIdx.x;  // 0..25599
  const int b = pix / HW_, hw = pix % HW_;
  const int h = hw / W_,  w  = hw % W_;
  float acc[18];
  #pragma unroll
  for (int o = 0; o < 18; ++o) acc[o] = 0.f;
  for (int c = 0; c < C_; ++c) {
    const float* xp = x + (b * C_ + c) * HW_;
    float xv[9];
    #pragma unroll
    for (int k = 0; k < 9; ++k) {
      const int yy = h + k / 3 - 1, xx = w + k % 3 - 1;
      const bool ok = (yy >= 0) && (yy < H_) && (xx >= 0) && (xx < W_);
      xv[k] = ok ? xp[yy * W_ + xx] : 0.f;
    }
    const float* wp = w_off + c * KK_;        // + o*C_*9 + k below (uniform)
    #pragma unroll
    for (int o = 0; o < 18; ++o) {
      #pragma unroll
      for (int k = 0; k < 9; ++k)
        acc[o] = fmaf(xv[k], wp[o * C_ * KK_ + k], acc[o]);
    }
  }
  #pragma unroll
  for (int o = 0; o < 18; ++o)
    offs[(b * 18 + o) * HW_ + hw] = acc[o];
}

// ---------------------------------------------------------------------------
// Kernel 4: main deformable GEMM.
// Workgroup: 256 threads (8 waves), 64-pixel M-tile x all 256 couts.
//  - Phase 0: bilinear coordinate table (64 pixels x 9 taps -> 4 addr+weight).
//  - Per K-chunk (32 channels): build bf16 A tile in LDS, then each wave does
//    4 M-subtiles x 2 N-tiles with v_wmma_f32_16x16x32_bf16, 9 K-steps.
//  Per K-step per wave: 8 WMMA vs 8 ds_load_b128 + 4 global_load_b128.
// ---------------------------------------------------------------------------
__global__ __launch_bounds__(256) void deform_gemm_kernel(
    const float* __restrict__ xt, const float* __restrict__ offs,
    const __bf16* __restrict__ bfrag, float* __restrict__ out)
{
  __shared__ __align__(16) __bf16 Atile[MTILE * KPAD];   // 37,888 B
  __shared__ int   cbase[ENT_][4];                       //  9,216 B
  __shared__ float cwt[ENT_][4];                         //  9,216 B

  const int tid   = threadIdx.x;
  const int pbase = blockIdx.x * MTILE;        // 64-pixel tile, never crosses b
  const int b     = pbase / HW_;
  const int hwb   = pbase % HW_;

  // --- Phase 0: bilinear coordinates & weights per (pixel, tap) -------------
  for (int e = tid; e < ENT_; e += 256) {
    const int m = e / KK_, kk = e % KK_;
    const int hw = hwb + m;
    const int h = hw / W_, w = hw % W_;
    const float dy = offs[(b * 18 + 2 * kk) * HW_ + hw];
    const float dx = offs[(b * 18 + 2 * kk + 1) * HW_ + hw];
    const float py = (float)(h + kk / 3 - 1) + dy;
    const float px = (float)(w + kk % 3 - 1) + dx;
    const float y0 = floorf(py), x0 = floorf(px);
    const float wy1 = py - y0, wx1 = px - x0;
    const float wy0 = 1.f - wy1, wx0 = 1.f - wx1;
    #pragma unroll
    for (int t = 0; t < 4; ++t) {
      const float ty = y0 + (float)(t >> 1);
      const float tx = x0 + (float)(t & 1);
      const bool ok = (ty >= 0.f) && (ty <= (float)(H_ - 1)) &&
                      (tx >= 0.f) && (tx <= (float)(W_ - 1));
      cbase[e][t] = ok ? ((b * H_ + (int)ty) * W_ + (int)tx) * C_ : 0;
      cwt[e][t]   = ok ? ((t >> 1) ? wy1 : wy0) * ((t & 1) ? wx1 : wx0) : 0.f;
    }
  }

  const int wave  = tid >> 5;
  const int lane  = tid & 31;
  const int lhalf = (lane & 16) ? 1 : 0;     // 0: lanes 0-15, 1: lanes 16-31
  const int lrow  = lane & 15;

  f32x8 acc[4][2] = {};                      // [m-subtile][n-tile-of-wave]

  for (int cc = 0; cc < NCHUNK; ++cc) {
    __syncthreads();  // coord table ready / previous GEMM done with Atile

    // --- Phase 1: build bf16 A chunk for channels [cc*32, cc*32+32) --------
    {
      const int cl = tid & 31;               // local channel (== lane)
      const int c  = cc * CCH + cl;
      const int eg = tid >> 5;               // 8 groups of 72 entries
      const float* xc = xt + c;
      for (int e = eg * 72; e < eg * 72 + 72; ++e) {
        const int m = e / KK_, kk = e % KK_;
        const float v = cwt[e][0] * xc[cbase[e][0]]
                      + cwt[e][1] * xc[cbase[e][1]]
                      + cwt[e][2] * xc[cbase[e][2]]
                      + cwt[e][3] * xc[cbase[e][3]];
        Atile[m * KPAD + cl * KK_ + kk] = (__bf16)v;
      }
    }
    __syncthreads();

    // --- Phase 2: WMMA GEMM over this K-chunk ------------------------------
    for (int ktl = 0; ktl < KTL; ++ktl) {
      const int kt = cc * KTL + ktl;         // global 32-wide K-tile index
      bf16x16 a[4];
      #pragma unroll
      for (int ms = 0; ms < 4; ++ms) {
        // 16-bit A layout (matches packed B): lanes 0-15 hold K {0..15}; 16-31 hold {16..31}
        const __bf16* ap = &Atile[(ms * 16 + lrow) * KPAD + ktl * 32 + lhalf * 16];
        const bf16x8 lo = *(const bf16x8*)ap;        // K k0..k0+7
        const bf16x8 hi = *(const bf16x8*)(ap + 8);  // K k0+8..k0+15
        a[ms] = __builtin_shufflevector(lo, hi,
                                        0, 1, 2, 3, 4, 5, 6, 7,
                                        8, 9, 10, 11, 12, 13, 14, 15);
      }
      #pragma unroll
      for (int j = 0; j < 2; ++j) {
        const int ntile = wave * 2 + j;
        const bf16x16 bm =
            *(const bf16x16*)(bfrag + (((kt * 16 + ntile) * 32 + lane) << 4));
        #pragma unroll
        for (int ms = 0; ms < 4; ++ms) {
          acc[ms][j] = __builtin_amdgcn_wmma_f32_16x16x32_bf16(
              false, a[ms], false, bm, (short)0, acc[ms][j], false, false);
        }
      }
    }
  }

  // --- Epilogue: D layout: VGPR v -> row M = v + 8*lhalf, col N = lane&15 ---
  #pragma unroll
  for (int ms = 0; ms < 4; ++ms) {
    #pragma unroll
    for (int j = 0; j < 2; ++j) {
      const int n = (wave * 2 + j) * 16 + lrow;
      float* op = out + (b * COUT_ + n) * HW_ + hwb + ms * 16 + lhalf * 8;
      #pragma unroll
      for (int v = 0; v < 8; ++v) op[v] = acc[ms][j][v];
    }
  }
}

// ---------------------------------------------------------------------------
// Host launch
// ---------------------------------------------------------------------------
extern "C" void kernel_launch(void* const* d_in, const int* in_sizes, int n_in,
                              void* d_out, int out_size, void* d_ws, size_t ws_size,
                              hipStream_t stream)
{
  (void)in_sizes; (void)n_in; (void)out_size; (void)ws_size;
  const float* x     = (const float*)d_in[0];   // [4,256,80,80]
  const float* w_off = (const float*)d_in[1];   // [18,256,3,3]
  const float* w_def = (const float*)d_in[2];   // [256,256,3,3]
  float* out = (float*)d_out;                   // [4,256,80,80]

  char* ws = (char*)d_ws;
  float*  xt    = (float*)(ws);                            // 26,214,400 B  (NHWC x)
  __bf16* bfrag = (__bf16*)(ws + 26214400);                //  1,179,648 B  (packed B)
  float*  offs  = (float*)(ws + 26214400 + 1179648);       //  1,843,200 B  (offsets)

  {
    dim3 tb(32, 8);
    dim3 tg(HW_ / 32, C_ / 32, B_);
    transpose_nchw_nhwc<<<tg, tb, 0, stream>>>(x, xt);
  }
  pack_bfrag<<<2304, 256, 0, stream>>>(w_def, bfrag);
  offset_conv_kernel<<<100, 256, 0, stream>>>(x, w_off, offs);
  deform_gemm_kernel<<<400, 256, 0, stream>>>(xt, offs, bfrag, out);
}
